// SingleHeadSelfAttention_21225728377037
// MI455X (gfx1250) — matrix operand order repaired and checked by reading, not compiled, not run on hardware
//
#include <hip/hip_runtime.h>
#include <hip/hip_bf16.h>

typedef _Float16 f16;
typedef __attribute__((ext_vector_type(16))) _Float16 v16h;
typedef __attribute__((ext_vector_type(8)))  _Float16 h8;
typedef __attribute__((ext_vector_type(8)))  float    v8f;
typedef __attribute__((ext_vector_type(4)))  float    f4v;
typedef __attribute__((ext_vector_type(4)))  unsigned int u32x4;
typedef __attribute__((ext_vector_type(8)))  int      i32x8;
typedef __attribute__((ext_vector_type(4)))  int      i32x4;

#define BATCH   4
#define SEQ     4096
#define DMODEL  512
#define SCALE   0.044194173824159216f   // 1/sqrt(512)
#define NEG_INF (-__builtin_inff())

#if defined(__has_builtin)
#  if __has_builtin(__builtin_amdgcn_tensor_load_to_lds) && __has_builtin(__builtin_amdgcn_s_wait_tensorcnt)
#    define USE_TDM 1
#  else
#    define USE_TDM 0
#  endif
#else
#  define USE_TDM 0
#endif

// ---------- helpers ----------
__device__ __forceinline__ v16h make_afrag(h8 lo, h8 hi) {
    return __builtin_shufflevector(lo, hi, 0,1,2,3,4,5,6,7,8,9,10,11,12,13,14,15);
}
__device__ __forceinline__ float hmax16(float v) {
    v = fmaxf(v, __shfl_xor(v, 1));
    v = fmaxf(v, __shfl_xor(v, 2));
    v = fmaxf(v, __shfl_xor(v, 4));
    v = fmaxf(v, __shfl_xor(v, 8));
    return v;
}
__device__ __forceinline__ float hsum16(float v) {
    v += __shfl_xor(v, 1);
    v += __shfl_xor(v, 2);
    v += __shfl_xor(v, 4);
    v += __shfl_xor(v, 8);
    return v;
}

// S-tile pair: two independent WMMA chains sharing one Q A-fragment stream from LDS
__device__ __forceinline__ void scores_pair(const f16* __restrict__ qs, int r, int hh,
                                            const f16* __restrict__ k0row,
                                            const f16* __restrict__ k1row,
                                            v8f& s0, v8f& s1)
{
    #pragma unroll 4
    for (int kk = 0; kk < DMODEL; kk += 32) {
        h8 lo = *(const h8*)(qs + r * 520 + kk + 16 * hh);
        h8 hi = *(const h8*)(qs + r * 520 + kk + 16 * hh + 8);
        v16h A  = make_afrag(lo, hi);
        v16h B0 = *(const v16h*)(k0row + kk + 16 * hh);
        v16h B1 = *(const v16h*)(k1row + kk + 16 * hh);
        s0 = __builtin_amdgcn_wmma_f32_16x16x32_f16(false, A, false, B0, (short)0, s0, false, false);
        s1 = __builtin_amdgcn_wmma_f32_16x16x32_f16(false, A, false, B1, (short)0, s1, false, false);
    }
}

// ---------- fp32 -> fp16 conversion ----------
__global__ void cvt_f32_f16(const float* __restrict__ src, f16* __restrict__ dst, int n) {
    int i = blockIdx.x * blockDim.x + threadIdx.x;
    int stride = gridDim.x * blockDim.x;
    for (; i < n; i += stride) dst[i] = (f16)src[i];
}

// ---------- zero-fill attention region ----------
__global__ void zero_f32v4(float* __restrict__ p, long long n4) {
    long long i = (long long)blockIdx.x * blockDim.x + threadIdx.x;
    long long stride = (long long)gridDim.x * blockDim.x;
    f4v z = {};
    for (; i < n4; i += stride) ((f4v*)p)[i] = z;
}

// ---------- QKV projection: Q = x Wq^T, K = x Wk^T, V = x Wv^T (V stored transposed) ----------
__global__ void __launch_bounds__(32)
qkv_kernel(const f16* __restrict__ xh,
           const f16* __restrict__ wq, const f16* __restrict__ wk, const f16* __restrict__ wv,
           f16* __restrict__ Qh, f16* __restrict__ Kh, f16* __restrict__ Vt)
{
    const int lane = threadIdx.x;
    const int hh   = lane >> 4;
    const int r    = lane & 15;
    const int bm   = blockIdx.x;             // b*(SEQ/16)+mt
    const int b    = bm >> 8;
    const int mt   = bm & 255;
    const int m0   = mt * 16;
    const int n0   = blockIdx.y * 16;

    v8f accQ = {}; v8f accK = {}; v8f accV = {};

    const f16* arow = xh + ((size_t)(b * SEQ + m0 + r)) * DMODEL;
    const f16* qrow = wq + (size_t)(n0 + r) * DMODEL;
    const f16* krow = wk + (size_t)(n0 + r) * DMODEL;
    const f16* vrow = wv + (size_t)(n0 + r) * DMODEL;

    #pragma unroll 4
    for (int kk = 0; kk < DMODEL; kk += 32) {
        h8 lo = *(const h8*)(arow + kk + 16 * hh);
        h8 hi = *(const h8*)(arow + kk + 16 * hh + 8);
        v16h A = make_afrag(lo, hi);
        v16h Bq = *(const v16h*)(qrow + kk + 16 * hh);
        v16h Bk = *(const v16h*)(krow + kk + 16 * hh);
        v16h Bv = *(const v16h*)(vrow + kk + 16 * hh);
        accQ = __builtin_amdgcn_wmma_f32_16x16x32_f16(false, A, false, Bq, (short)0, accQ, false, false);
        accK = __builtin_amdgcn_wmma_f32_16x16x32_f16(false, A, false, Bk, (short)0, accK, false, false);
        accV = __builtin_amdgcn_wmma_f32_16x16x32_f16(false, A, false, Bv, (short)0, accV, false, false);
    }

    #pragma unroll
    for (int e = 0; e < 8; ++e) {
        const int row = m0 + e + 8 * hh;     // sequence position
        const int col = n0 + r;              // output feature
        Qh[((size_t)(b * SEQ + row)) * DMODEL + col] = (f16)accQ[e];
        Kh[((size_t)(b * SEQ + row)) * DMODEL + col] = (f16)accK[e];
        Vt[((size_t)b * DMODEL + col) * SEQ + row]   = (f16)accV[e];  // transposed [b][d][s]
    }
}

// ---------- causal flash attention, two-pass, writes attention matrix + residual output ----------
__global__ void __launch_bounds__(32)
attn_kernel(const float* __restrict__ x,
            const f16* __restrict__ Qh, const f16* __restrict__ Kh, const f16* __restrict__ Vt,
            float* __restrict__ out, float* __restrict__ attn)
{
    __shared__ f16 qs[16 * 520];   // Q tile 16 x 512, padded pitch (matches TDM pad: 8 halves/row)
    __shared__ f16 ps[16 * 40];    // P tile 16 x 32, padded pitch

    const int lane = threadIdx.x;
    const int hh   = lane >> 4;
    const int r    = lane & 15;
    const int qt   = blockIdx.x;
    const int b    = blockIdx.y;
    const int q0   = qt * 16;

    // ---- stage Q tile into LDS ----
    const f16* qg = Qh + ((size_t)(b * SEQ + q0)) * DMODEL;
#if USE_TDM
    {
        // Tensor DMA: 16x512 tile of 2-byte elements, row stride 512, LDS pad 4 DWORDs
        // every 256 DWORDs (one row) -> LDS pitch = 520 halves.
        const unsigned lds_off = (unsigned)(uintptr_t)(void*)qs;   // LDS aperture: low 32 bits
        const unsigned long long ga = (unsigned long long)(uintptr_t)qg;
        u32x4 g0 = {
            1u,                                                    // count=1, user mode
            lds_off,                                               // D#.lds_addr
            (unsigned)(ga & 0xFFFFFFFFull),                        // global_addr[31:0]
            (unsigned)((ga >> 32) & 0x1FFFFFFull) | (2u << 30)     // global_addr[56:32] | type=2
        };
        i32x8 g1 = {
            (int)((1u << 16) | (1u << 20) | (7u << 22) | (3u << 25)), // data_size=2B, pad_en,
                                                                      // pad_interval=256dw, pad_amount=4dw
            (int)(512u << 16),      // tensor_dim0 = 512 (low 16 bits at bit 48)
            (int)(16u << 16),       // tensor_dim0 hi = 0 | tensor_dim1 = 16
            (int)(512u << 16),      // tensor_dim1 hi = 0 | tile_dim0 = 512
            16,                     // tile_dim1 = 16, tile_dim2 = 0
            512,                    // tensor_dim0_stride low 32 = 512 elements
            0, 0                    // stride hi / tensor_dim1_stride = 0 (2D)
        };
        i32x4 gz = {0, 0, 0, 0};
#if __clang_major__ >= 23
        i32x8 gz8 = {0, 0, 0, 0, 0, 0, 0, 0};
        __builtin_amdgcn_tensor_load_to_lds(g0, g1, gz, gz, gz8, 0);
#else
        __builtin_amdgcn_tensor_load_to_lds(g0, g1, gz, gz, 0);
#endif
        __builtin_amdgcn_s_wait_tensorcnt(0);
    }
#else
    for (int i = lane; i < 16 * 64; i += 32) {
        int row = i >> 6, c8 = i & 63;
        *(h8*)(qs + row * 520 + c8 * 8) = *(const h8*)(qg + (size_t)row * DMODEL + c8 * 8);
    }
#endif
    __syncthreads();

    float m[8], l[8];
    #pragma unroll
    for (int e = 0; e < 8; ++e) { m[e] = NEG_INF; l[e] = 0.f; }

    const int nkt   = qt + 1;          // causal: key tiles 0..qt
    const int npair = (nkt + 1) >> 1;  // key-tile pairs (last may be half-masked)
    const f16* kbat = Kh + ((size_t)(b * SEQ + r)) * DMODEL;   // per-lane K row base

    // ---- pass 1: row max / row sum, two independent WMMA chains per pair ----
    for (int kp = 0; kp < npair; ++kp) {
        const f16* k0row = kbat + (size_t)(kp * 32) * DMODEL;
        const f16* k1row = k0row + (size_t)16 * DMODEL;
        if (kp + 1 < npair) {   // L2 prefetch of next pair's K rows (global_prefetch_b8)
            __builtin_prefetch(k0row + (size_t)32 * DMODEL, 0, 1);
            __builtin_prefetch(k1row + (size_t)32 * DMODEL, 0, 1);
        }
        v8f s0 = {}, s1 = {};
        scores_pair(qs, r, hh, k0row, k1row, s0, s1);

        #pragma unroll
        for (int e = 0; e < 8; ++e) {
            const int qg2 = q0 + e + 8 * hh;
            const int kg0 = kp * 32 + r;
            const int kg1 = kg0 + 16;
            float sv0 = s0[e] * SCALE; if (kg0 > qg2) sv0 = NEG_INF;
            float sv1 = s1[e] * SCALE; if (kg1 > qg2) sv1 = NEG_INF;
            float nm = fmaxf(m[e], hmax16(fmaxf(sv0, sv1)));
            float p  = __expf(sv0 - nm) + __expf(sv1 - nm);
            l[e] = l[e] * __expf(m[e] - nm) + hsum16(p);
            m[e] = nm;
        }
    }
    float il[8];
    #pragma unroll
    for (int e = 0; e < 8; ++e) il[e] = 1.0f / l[e];

    // ---- pass 2: write normalized P, accumulate O = P @ V ----
    v8f o[32];
    #pragma unroll
    for (int t = 0; t < 32; ++t) o[t] = (v8f){};

    for (int kp = 0; kp < npair; ++kp) {
        const int kbase = kp * 32;
        const f16* k0row = kbat + (size_t)kbase * DMODEL;
        const f16* k1row = k0row + (size_t)16 * DMODEL;

        v8f s0 = {}, s1 = {};
        scores_pair(qs, r, hh, k0row, k1row, s0, s1);

        #pragma unroll
        for (int e = 0; e < 8; ++e) {
            const int qg2 = q0 + e + 8 * hh;
            const int kg0 = kbase + r;
            const int kg1 = kg0 + 16;
            float sv0 = s0[e] * SCALE; if (kg0 > qg2) sv0 = NEG_INF;
            float sv1 = s1[e] * SCALE; if (kg1 > qg2) sv1 = NEG_INF;
            const float p0 = __expf(sv0 - m[e]) * il[e];   // 0 for masked entries
            const float p1 = __expf(sv1 - m[e]) * il[e];
            attn[((size_t)(b * SEQ + qg2)) * SEQ + kg0] = p0;
            attn[((size_t)(b * SEQ + qg2)) * SEQ + kg1] = p1;
            ps[(e + 8 * hh) * 40 + r]      = (f16)p0;
            ps[(e + 8 * hh) * 40 + 16 + r] = (f16)p1;
        }
        __syncthreads();   // LDS C-layout -> A-layout relayout of P

        h8 plo = *(const h8*)(ps + r * 40 + 16 * hh);
        h8 phi = *(const h8*)(ps + r * 40 + 16 * hh + 8);
        v16h PA = make_afrag(plo, phi);

        #pragma unroll
        for (int dchunk = 0; dchunk < 32; ++dchunk) {
            const f16* vrow = Vt + ((size_t)b * DMODEL + dchunk * 16 + r) * SEQ + kbase + 16 * hh;
            v16h B = *(const v16h*)vrow;
            o[dchunk] = __builtin_amdgcn_wmma_f32_16x16x32_f16(false, PA, false, B, (short)0, o[dchunk], false, false);
        }
    }

    // ---- epilogue: out = x + O ----
    #pragma unroll
    for (int dchunk = 0; dchunk < 32; ++dchunk) {
        #pragma unroll
        for (int e = 0; e < 8; ++e) {
            const int qg2 = q0 + e + 8 * hh;
            const int dc  = dchunk * 16 + r;
            const size_t idx = ((size_t)(b * SEQ + qg2)) * DMODEL + dc;
            out[idx] = x[idx] + o[dchunk][e];
        }
    }
}

extern "C" void kernel_launch(void* const* d_in, const int* in_sizes, int n_in,
                              void* d_out, int out_size, void* d_ws, size_t ws_size,
                              hipStream_t stream)
{
    (void)in_sizes; (void)n_in; (void)out_size; (void)ws_size;

    const float* x  = (const float*)d_in[0];
    const float* wq = (const float*)d_in[1];
    const float* wk = (const float*)d_in[2];
    const float* wv = (const float*)d_in[3];

    char* ws = (char*)d_ws;
    size_t off = 0;
    const size_t xElems = (size_t)BATCH * SEQ * DMODEL;   // 8,388,608
    const size_t wElems = (size_t)DMODEL * DMODEL;        // 262,144

    f16* xh  = (f16*)(ws + off); off += xElems * 2;
    f16* wqh = (f16*)(ws + off); off += wElems * 2;
    f16* wkh = (f16*)(ws + off); off += wElems * 2;
    f16* wvh = (f16*)(ws + off); off += wElems * 2;
    f16* Qh  = (f16*)(ws + off); off += xElems * 2;
    f16* Kh  = (f16*)(ws + off); off += xElems * 2;
    f16* Vt  = (f16*)(ws + off); off += xElems * 2;       // ~68.7 MB total

    float* out  = (float*)d_out;
    float* attn = out + xElems;

    cvt_f32_f16<<<4096, 256, 0, stream>>>(x,  xh,  (int)xElems);
    cvt_f32_f16<<<512,  256, 0, stream>>>(wq, wqh, (int)wElems);
    cvt_f32_f16<<<512,  256, 0, stream>>>(wk, wkh, (int)wElems);
    cvt_f32_f16<<<512,  256, 0, stream>>>(wv, wvh, (int)wElems);

    qkv_kernel<<<dim3(BATCH * SEQ / 16, DMODEL / 16), 32, 0, stream>>>(
        xh, wqh, wkh, wvh, Qh, Kh, Vt);

    zero_f32v4<<<8192, 256, 0, stream>>>(attn, (long long)BATCH * SEQ * SEQ / 4);

    attn_kernel<<<dim3(SEQ / 16, BATCH), 32, 0, stream>>>(x, Qh, Kh, Vt, out, attn);
}
